// SelfAttention_48112223650211
// MI455X (gfx1250) — matrix operand order repaired and checked by reading, not compiled, not run on hardware
//
#include <hip/hip_runtime.h>
#include <hip/hip_bf16.h>

// CDNA5 / gfx1250 WMMA types
typedef __attribute__((ext_vector_type(8)))  __bf16 v8bf;
typedef __attribute__((ext_vector_type(16))) __bf16 v16bf;
typedef __attribute__((ext_vector_type(8)))  float  v8f;
// TDM descriptor groups
typedef __attribute__((ext_vector_type(4))) unsigned tdm_v4u;
typedef __attribute__((ext_vector_type(8))) int      tdm_v8i;
typedef __attribute__((ext_vector_type(4))) int      tdm_v4i;

#define B_   2
#define S_   2048
#define H_   16
#define D_   64
#define E_   1024
// (1/sqrt(E)) * log2(e): scores kept in log2 domain so exp2 == v_exp_f32 directly
#define SCALE_LOG2E 0.0450842117f

__device__ __forceinline__ v16bf cat8(v8bf lo, v8bf hi) {
    return __builtin_shufflevector(lo, hi, 0, 1, 2, 3, 4, 5, 6, 7,
                                   8, 9, 10, 11, 12, 13, 14, 15);
}
// B fragment: lane = col, 16 contiguous bf16 K-values at p (16B aligned)
__device__ __forceinline__ v16bf ldB(const __bf16* p) {
    return cat8(*(const v8bf*)p, *(const v8bf*)(p + 8));
}
// A fragment: lane = row; 16 contiguous K-values: K = 16*hi+e
__device__ __forceinline__ v16bf ldA(const __bf16* p, int hi) {
    return cat8(*(const v8bf*)(p + 16 * hi), *(const v8bf*)(p + 16 * hi + 8));
}
__device__ __forceinline__ v8bf cvt8(float4 a, float4 b) {
    v8bf r;
    r[0] = (__bf16)a.x; r[1] = (__bf16)a.y; r[2] = (__bf16)a.z; r[3] = (__bf16)a.w;
    r[4] = (__bf16)b.x; r[5] = (__bf16)b.y; r[6] = (__bf16)b.z; r[7] = (__bf16)b.w;
    return r;
}
__device__ __forceinline__ v8f wmma_bf16(v16bf a, v16bf b, v8f c) {
    return __builtin_amdgcn_wmma_f32_16x16x32_bf16(false, a, false, b, (short)0, c, false, false);
}
__device__ __forceinline__ v8f vzero8() {
    v8f z = {0.f, 0.f, 0.f, 0.f, 0.f, 0.f, 0.f, 0.f};
    return z;
}

// ---------------------------------------------------------------------------
// TDM: load a 64(rows) x 64(bf16) tile, row stride 1024 elems, into LDS with
// hardware padding of 4 DWORDs after every 32 DWORDs (-> 72-element LDS rows).
// D# layout per CDNA5 ISA (group0 128b, group1 256b).
// This toolchain exposes the 6-arg builtin:
//   (uint32x4 g0, int32x8 g1, int32x4 g2, int32x4 g3, int32x8, i32 cpol)
// ---------------------------------------------------------------------------
#if __has_builtin(__builtin_amdgcn_tensor_load_to_lds)
#define HAVE_TDM 1
__device__ __forceinline__ void tdm_load_tile64(unsigned lds_byte, const void* gptr) {
    unsigned long long ga = (unsigned long long)gptr;
    tdm_v4u g0;
    g0[0] = 1u;                                     // count=1, user descriptor
    g0[1] = lds_byte;                               // lds_addr
    g0[2] = (unsigned)(ga & 0xffffffffu);           // global_addr[31:0]
    g0[3] = (unsigned)((ga >> 32) & 0x01ffffffu)    // global_addr[56:32]
          | (2u << 30);                             // type=2 (image)
    tdm_v8i g1;
    g1[0] = (1 << 16)     // data_size = 1 -> 2 bytes
          | (1 << 20)     // pad_enable
          | (4 << 22)     // pad_interval code 4 -> every 32 DWORDs
          | (3 << 25);    // pad_amount  code 3 -> 4 DWORDs (8 bf16)
    g1[1] = (D_ & 0xffff) << 16;                    // tensor_dim0[15:0]=64
    g1[2] = ((D_ >> 16) & 0xffff) | ((S_ & 0xffff) << 16);  // dim0 hi | dim1 lo = 2048
    g1[3] = ((S_ >> 16) & 0xffff) | (D_ << 16);     // dim1 hi | tile_dim0 = 64
    g1[4] = 64;                                     // tile_dim1 = 64 rows, tile_dim2 = 0
    g1[5] = H_ * D_;                                // tensor_dim0_stride = 1024 elems
    g1[6] = 0;                                      // stride hi | dim1_stride lo (unused, 2D)
    g1[7] = 0;
    tdm_v4i z4 = {0, 0, 0, 0};
    tdm_v8i z8 = {0, 0, 0, 0, 0, 0, 0, 0};
    __builtin_amdgcn_tensor_load_to_lds(g0, g1, z4, z4, z8, 0);
}
#else
#define HAVE_TDM 0
#endif

// ---------------------------------------------------------------------------
// Kernel 1: per-head projection  dst[row, :] = x[row, :] @ W^T  (rows = B*S*H)
// ---------------------------------------------------------------------------
__global__ void __launch_bounds__(128)
proj_kernel(const float* __restrict__ xq, const float* __restrict__ xk,
            const float* __restrict__ xv,
            const float* __restrict__ Wq, const float* __restrict__ Wk,
            const float* __restrict__ Wv,
            __bf16* __restrict__ qb, __bf16* __restrict__ kb, __bf16* __restrict__ vb) {
    const float* x; const float* W; __bf16* dst;
    if (blockIdx.y == 0)      { x = xq; W = Wq; dst = qb; }
    else if (blockIdx.y == 1) { x = xk; W = Wk; dst = kb; }
    else                      { x = xv; W = Wv; dst = vb; }

    __shared__ __align__(16) __bf16 Wl[64 * 72];   // Wl[n][k] = bf16(W[n][k])
    const int tid = threadIdx.x;
    for (int idx = tid; idx < 512; idx += 128) {
        int n = idx >> 3, c = idx & 7;
        const float4* src = (const float4*)(W + n * 64 + c * 8);
        *(v8bf*)&Wl[n * 72 + c * 8] = cvt8(src[0], src[1]);
    }
    __syncthreads();

    const int wave = tid >> 5, lane = tid & 31;
    const int col = lane & 15, hi = lane >> 4;
    const int rowbase = blockIdx.x * 64 + wave * 16;

    v16bf a[2];
    const float* xrow = x + (rowbase + col) * 64;
#pragma unroll
    for (int ks = 0; ks < 2; ++ks) {
        const float4* r0 = (const float4*)(xrow + 32 * ks + 16 * hi);
        const float4* r1 = (const float4*)(xrow + 32 * ks + 16 * hi + 8);
        a[ks] = cat8(cvt8(r0[0], r0[1]), cvt8(r1[0], r1[1]));
    }

#pragma unroll
    for (int t = 0; t < 4; ++t) {
        v8f c = vzero8();
#pragma unroll
        for (int ks = 0; ks < 2; ++ks)
            c = wmma_bf16(a[ks], ldB(&Wl[(16 * t + col) * 72 + 32 * ks + 16 * hi]), c);
#pragma unroll
        for (int i = 0; i < 8; ++i)
            dst[(rowbase + i + 8 * hi) * 64 + 16 * t + col] = (__bf16)c[i];
    }
}

// ---------------------------------------------------------------------------
// Kernel 2: flash attention. grid.x = S/64 q-tiles, grid.y = B*H.
// K tile loaded by the Tensor Data Mover (padded rows), V transposed manually,
// P restaged per-wave for C-layout -> A-layout; softmax in log2 domain.
// ---------------------------------------------------------------------------
__global__ void __launch_bounds__(128)
flash_kernel(const __bf16* __restrict__ qb, const __bf16* __restrict__ kb,
             const __bf16* __restrict__ vb, __bf16* __restrict__ ob) {
    const int bh = blockIdx.y;
    const int b = bh >> 4, h = bh & 15;
    const int tid = threadIdx.x, wave = tid >> 5, lane = tid & 31;
    const int col = lane & 15, hi = lane >> 4;
    const int q0 = blockIdx.x * 64 + wave * 16;

    __shared__ __align__(16) __bf16 Kl[64 * 72];      // [key][d]
    __shared__ __align__(16) __bf16 Vt[64 * 72];      // [d][key]  (transposed)
    __shared__ __align__(16) __bf16 Pl[4][16 * 72];   // per-wave P tile [row][key]

    const __bf16* qrow = qb + ((b * S_ + q0 + col) * H_ + h) * D_;
    v16bf aq[2];
#pragma unroll
    for (int ks = 0; ks < 2; ++ks)
        aq[ks] = ldA(qrow + 32 * ks, hi);

    v8f acc[4];
    float m[8], l[8];
#pragma unroll
    for (int t = 0; t < 4; ++t) acc[t] = vzero8();
#pragma unroll
    for (int i = 0; i < 8; ++i) { m[i] = -3.0e38f; l[i] = 0.f; }

    for (int c0 = 0; c0 < S_; c0 += 64) {
        __syncthreads();   // previous chunk fully consumed before overwrite
        const __bf16* kbase = kb + ((b * S_ + c0) * H_ + h) * D_;
        const __bf16* vbase = vb + ((b * S_ + c0) * H_ + h) * D_;
#if HAVE_TDM
        if (wave == 0)
            tdm_load_tile64((unsigned)(size_t)&Kl[0], kbase);
#else
        for (int idx = tid; idx < 512; idx += 128) {
            int r = idx >> 3, c8 = idx & 7;
            *(v8bf*)&Kl[r * 72 + c8 * 8] = *(const v8bf*)(kbase + r * (H_ * D_) + c8 * 8);
        }
#endif
        // V: vectorized global load, transpose-store into Vt[d][key]
        for (int idx = tid; idx < 512; idx += 128) {
            int r = idx >> 3, c8 = idx & 7;
            v8bf vv = *(const v8bf*)(vbase + r * (H_ * D_) + c8 * 8);
#pragma unroll
            for (int j = 0; j < 8; ++j)
                Vt[(c8 * 8 + j) * 72 + r] = vv[j];
        }
#if HAVE_TDM
        if (wave == 0)
            __builtin_amdgcn_s_wait_tensorcnt(0);
#endif
        __syncthreads();

        // scores (log2 domain): S2 = (Q @ K^T) * (1/sqrt(E)) * log2(e)
        v8f s[4];
#pragma unroll
        for (int t = 0; t < 4; ++t) {
            v8f c = vzero8();
#pragma unroll
            for (int ks = 0; ks < 2; ++ks)
                c = wmma_bf16(aq[ks], ldB(&Kl[(16 * t + col) * 72 + 32 * ks + 16 * hi]), c);
#pragma unroll
            for (int i = 0; i < 8; ++i) s[t][i] = c[i] * SCALE_LOG2E;
        }

        // online softmax (exp2): row i+8*hi lives within a 16-lane half
#pragma unroll
        for (int i = 0; i < 8; ++i) {
            float rmax = fmaxf(fmaxf(s[0][i], s[1][i]), fmaxf(s[2][i], s[3][i]));
#pragma unroll
            for (int mk = 1; mk < 16; mk <<= 1)
                rmax = fmaxf(rmax, __shfl_xor(rmax, mk));
            float mn = fmaxf(m[i], rmax);
            float corr = exp2f(m[i] - mn);
            m[i] = mn;
            float rsum = 0.f;
#pragma unroll
            for (int t = 0; t < 4; ++t) {
                float p = exp2f(s[t][i] - mn);
                rsum += p;
                Pl[wave][(i + 8 * hi) * 72 + 16 * t + col] = (__bf16)p;
            }
#pragma unroll
            for (int mk = 1; mk < 16; mk <<= 1)
                rsum += __shfl_xor(rsum, mk);
            l[i] = l[i] * corr + rsum;
#pragma unroll
            for (int t = 0; t < 4; ++t) acc[t][i] *= corr;
        }

        // O += P @ V
#pragma unroll
        for (int ks = 0; ks < 2; ++ks) {
            v16bf ap = ldA(&Pl[wave][col * 72 + 32 * ks], hi);
#pragma unroll
            for (int t = 0; t < 4; ++t)
                acc[t] = wmma_bf16(ap, ldB(&Vt[(16 * t + col) * 72 + 32 * ks + 16 * hi]), acc[t]);
        }
    }

    // normalize and store attention output [B,S,H,D] as bf16
#pragma unroll
    for (int i = 0; i < 8; ++i) {
        float inv = 1.f / l[i];
        int row = (b * S_ + q0 + i + 8 * hi) * H_ + h;
#pragma unroll
        for (int t = 0; t < 4; ++t)
            ob[row * D_ + 16 * t + col] = (__bf16)(acc[t][i] * inv);
    }
}

// ---------------------------------------------------------------------------
// Kernel 3: out = attn [4096 x 1024] @ Wo^T, fp32 output.
// ---------------------------------------------------------------------------
__global__ void __launch_bounds__(128)
outproj_kernel(const __bf16* __restrict__ ab, const float* __restrict__ Wo,
               float* __restrict__ out) {
    const int mt = blockIdx.x >> 4, nt = blockIdx.x & 15;
    const int m0 = mt * 64, n0 = nt * 64;
    const int tid = threadIdx.x, wave = tid >> 5, lane = tid & 31;
    const int col = lane & 15, hi = lane >> 4;

    __shared__ __align__(16) __bf16 Bl[64 * 40];   // Bl[n][k] = Wo[n0+n][kb+k], k<32

    v8f acc[4];
#pragma unroll
    for (int t = 0; t < 4; ++t) acc[t] = vzero8();

    for (int kb = 0; kb < E_; kb += 32) {
        __syncthreads();
        for (int idx = tid; idx < 256; idx += 128) {
            int n = idx >> 2, c = idx & 3;
            const float4* src = (const float4*)(Wo + (n0 + n) * E_ + kb + c * 8);
            *(v8bf*)&Bl[n * 40 + c * 8] = cvt8(src[0], src[1]);
        }
        __syncthreads();

        v16bf a = ldA(ab + (m0 + wave * 16 + col) * E_ + kb, hi);
#pragma unroll
        for (int t = 0; t < 4; ++t)
            acc[t] = wmma_bf16(a, ldB(&Bl[(16 * t + col) * 40 + 16 * hi]), acc[t]);
    }

#pragma unroll
    for (int t = 0; t < 4; ++t)
#pragma unroll
        for (int i = 0; i < 8; ++i)
            out[(m0 + wave * 16 + i + 8 * hi) * E_ + n0 + 16 * t + col] = acc[t][i];
}

// ---------------------------------------------------------------------------
extern "C" void kernel_launch(void* const* d_in, const int* in_sizes, int n_in,
                              void* d_out, int out_size, void* d_ws, size_t ws_size,
                              hipStream_t stream) {
    const float* query = (const float*)d_in[0];
    const float* key   = (const float*)d_in[1];
    const float* value = (const float*)d_in[2];
    const float* Wq    = (const float*)d_in[3];
    const float* Wk    = (const float*)d_in[4];
    const float* Wv    = (const float*)d_in[5];
    const float* Wo    = (const float*)d_in[6];
    float* out = (float*)d_out;

    const size_t rows = (size_t)B_ * S_ * H_;        // 65536
    __bf16* qb = (__bf16*)d_ws;                      // rows*64 bf16
    __bf16* kb = qb + rows * D_;
    __bf16* vb = kb + rows * D_;
    __bf16* ob = vb + rows * D_;                     // B*S x E bf16
    (void)in_sizes; (void)n_in; (void)out_size; (void)ws_size;

    dim3 gridP(1024, 3);
    proj_kernel<<<gridP, 128, 0, stream>>>(query, key, value, Wq, Wk, Wv, qb, kb, vb);

    dim3 gridF(S_ / 64, B_ * H_);
    flash_kernel<<<gridF, 128, 0, stream>>>(qb, kb, vb, ob);

    dim3 gridO((B_ * S_ / 64) * (E_ / 64));
    outproj_kernel<<<gridO, 128, 0, stream>>>(ob, Wo, out);
}
